// SO2_Attention_45586782880339
// MI455X (gfx1250) — matrix operand order repaired and checked by reading, not compiled, not run on hardware
//
#include <hip/hip_runtime.h>
#include <hip/hip_bf16.h>

#define N_NODES 50000
#define N_EDGES 500000
#define FEAT    240

typedef __attribute__((ext_vector_type(16))) _Float16 v16h;
typedef __attribute__((ext_vector_type(8)))  float    v8f;

// fast silu: x * rcp(1+exp(-x))  -> v_exp_f32 + v_rcp_f32 + v_add + v_mul
__device__ __forceinline__ float fast_silu(float x) {
    float e = __expf(-x);
    return x * __builtin_amdgcn_rcpf(1.f + e);
}

// ---------------------------------------------------------------------------
// Kernel 0: repack weights (f32 -> f16) into WMMA B-operand tile layout.
// B tile = 32(K) x 16(N): lane = n + 16*((k>>3)&1), half-slot s = (k&7) + 8*(k>>4).
// Tile table (512 halves each):
//   [0,16)   Ws0  (64x128):  tile = nt*2  + kt   (nt 0..7, kt 0..1)
//   [16,40)  Ws1  (96x128):  tile = 16 + nt*3 + kt
//   [40,64)  Ws2  (80x128 zero-padded to 96): tile = 40 + nt*3 + kt
//   [64,256) Wf1  (384x256): tile = 64 + nt*12 + kt  (nt 0..15)
//   [256,320)Wf2  (256x128): tile = 256 + nt*8 + kt
// ---------------------------------------------------------------------------
__global__ void prep_weights(const float* __restrict__ Ws0, const float* __restrict__ Ws1,
                             const float* __restrict__ Ws2, const float* __restrict__ Wf1,
                             const float* __restrict__ Wf2, _Float16* __restrict__ wsB) {
    int gid = blockIdx.x * 256 + threadIdx.x;
    if (gid >= 320 * 512) return;
    int tile = gid >> 9;
    int idx  = gid & 511;
    int lane = idx >> 4;
    int s    = idx & 15;
    int nl   = lane & 15;
    int kl   = (s & 7) + ((lane >> 4) << 3) + ((s >> 3) << 4);   // 0..31 within tile
    float v = 0.f;
    if (tile < 16) {
        int nt = tile >> 1, kt = tile & 1;
        v = Ws0[(kt * 32 + kl) * 128 + nt * 16 + nl];
    } else if (tile < 40) {
        int t = tile - 16, nt = t / 3, kt = t % 3;
        v = Ws1[(kt * 32 + kl) * 128 + nt * 16 + nl];
    } else if (tile < 64) {
        int t = tile - 40, nt = t / 3, kt = t % 3;
        int k = kt * 32 + kl;
        v = (k < 80) ? Ws2[k * 128 + nt * 16 + nl] : 0.f;
    } else if (tile < 256) {
        int t = tile - 64, nt = t / 12, kt = t % 12;
        v = Wf1[(kt * 32 + kl) * 256 + nt * 16 + nl];
    } else {
        int t = tile - 256, nt = t >> 3, kt = t & 7;
        v = Wf2[(kt * 32 + kl) * 128 + nt * 16 + nl];
    }
    wsB[(tile << 9) + idx] = (_Float16)v;
}

// ---------------------------------------------------------------------------
// Kernel 1: per-node e3nn linear for both center (Wc*) and neigh (Wn*) stacks.
// One thread per (node, output-channel). Outputs f32 rows of 240.
// ---------------------------------------------------------------------------
__global__ void node_linear(const float* __restrict__ x,
                            const float* __restrict__ Wc0, const float* __restrict__ bc0,
                            const float* __restrict__ Wc1, const float* __restrict__ Wc2,
                            const float* __restrict__ Wn0, const float* __restrict__ bn0,
                            const float* __restrict__ Wn1, const float* __restrict__ Wn2,
                            float* __restrict__ outC, float* __restrict__ outN) {
    long gid = (long)blockIdx.x * 256 + threadIdx.x;
    if (gid >= (long)N_NODES * FEAT) return;
    int node = (int)(gid / FEAT);
    int c    = (int)(gid % FEAT);
    const float* xr = x + (long)node * FEAT;
    float aC, aN;
    if (c < 64) {
        aC = bc0[c]; aN = bn0[c];
        #pragma unroll 4
        for (int i = 0; i < 64; ++i) {
            float xv = xr[i];
            aC += xv * Wc0[i * 64 + c] * 0.125f;
            aN += xv * Wn0[i * 64 + c] * 0.125f;
        }
    } else if (c < 160) {
        int cc = c - 64, k = cc / 3, d = cc % 3;
        const float sc = 0.17677669529663687f;  // 1/sqrt(32)
        aC = 0.f; aN = 0.f;
        #pragma unroll 4
        for (int m = 0; m < 32; ++m) {
            float xv = xr[64 + m * 3 + d];
            aC += xv * Wc1[m * 32 + k] * sc;
            aN += xv * Wn1[m * 32 + k] * sc;
        }
    } else {
        int cc = c - 160, k = cc / 5, d = cc % 5;
        aC = 0.f; aN = 0.f;
        #pragma unroll
        for (int m = 0; m < 16; ++m) {
            float xv = xr[160 + m * 5 + d];
            aC += xv * Wc2[m * 16 + k] * 0.25f;
            aN += xv * Wn2[m * 16 + k] * 0.25f;
        }
    }
    outC[gid] = aC;
    outN[gid] = aN;
}

// ---------------------------------------------------------------------------
// Kernel 2: edge pipeline. 256 threads = 8 wave32, 64 edges/block.
//   Step A : gather + Wigner rotations + sim, write f16 A-layout into LDS
//   GEMM1  : sim(240p256) -> 384 (block-diagonal Ws0/1/2), silu  -> LDS fused
//   GEMM2  : fused(384) -> 256 (Wf1), silu                       -> LDS h
//   GEMM3  : h(256) -> 128 (Wf2)                                 -> global out
// A tile (16x32 f16): lane = m + 16*((k>>3)&1), slot = (k&7) + 8*(k>>4).
// D tile (16x16 f32): vgpr r, m = r + 8*(lane>>4), n = lane&15.
// All per-wave tile indices go through readfirstlane so WMMA control flow is
// scalar (no exec masking around v_wmma; EXEC stays all-ones).
// ---------------------------------------------------------------------------
__launch_bounds__(256)
__global__ void edge_kernel(const int* __restrict__ eidx, const float* __restrict__ wig,
                            const float* __restrict__ wsC, const float* __restrict__ wsN,
                            const _Float16* __restrict__ wsB,
                            const float* __restrict__ bs0, const float* __restrict__ bs1,
                            const float* __restrict__ bs2, const float* __restrict__ bf1,
                            const float* __restrict__ bf2, float* __restrict__ out) {
    __shared__ __align__(32) _Float16 s_fused[4 * 12 * 512];  // 48 KB: fused A (K=384)
    __shared__ __align__(32) _Float16 s_ovl[4 * 8 * 512];     // 32 KB: sim A, then h A

    const int tid  = threadIdx.x;
    const int lane = tid & 31;
    const int wave = __builtin_amdgcn_readfirstlane(tid >> 5);  // scalar wave id
    const long e0  = (long)blockIdx.x * 64;

    // zero sim buffer (covers K padding 240..255 and tail edges)
    for (int i = tid; i < 4 * 8 * 512 / 2; i += 256)
        ((unsigned*)s_ovl)[i] = 0u;
    __syncthreads();

    // ---------------- Step A: gather + rotate + sim -> LDS (A layout) -------
    {
        const int el   = tid >> 2;   // edge within block, 0..63
        const int part = tid & 3;
        const long e   = e0 + el;
        if (e < N_EDGES) {
            const int mt = el >> 4, mm = el & 15;
            const int ci = eidx[e];
            const int ni = eidx[N_EDGES + e];
            const float* cf = wsC + (long)ci * FEAT;
            const float* nf = wsN + (long)ni * FEAT;
#define PUTSIM(K, V)                                                               \
    do {                                                                           \
        int _k = (K); int _kt = _k >> 5, _kk = _k & 31;                            \
        s_ovl[((mt * 8 + _kt) << 9) + ((mm + (((_kk >> 3) & 1) << 4)) << 4) +      \
              (_kk & 7) + ((_kk >> 4) << 3)] = (_Float16)(V);                      \
    } while (0)
            if (part == 0) {
                #pragma unroll 8
                for (int i = 0; i < 64; ++i) PUTSIM(i, cf[i] * nf[i]);
            } else if (part < 3) {
                const float* wb = wig + e * 81;
                float R[3][3];
                #pragma unroll
                for (int d = 0; d < 3; ++d)
                    #pragma unroll
                    for (int q = 0; q < 3; ++q) R[d][q] = wb[(1 + d) * 9 + 1 + q];
                const int m0 = (part - 1) * 16;
                for (int m = m0; m < m0 + 16; ++m) {
                    float c0 = cf[64 + m * 3], c1 = cf[65 + m * 3], c2 = cf[66 + m * 3];
                    float n0 = nf[64 + m * 3], n1 = nf[65 + m * 3], n2 = nf[66 + m * 3];
                    #pragma unroll
                    for (int q = 0; q < 3; ++q) {
                        float cr = c0 * R[0][q] + c1 * R[1][q] + c2 * R[2][q];
                        float nr = n0 * R[0][q] + n1 * R[1][q] + n2 * R[2][q];
                        PUTSIM(64 + m * 3 + q, cr * nr);
                    }
                }
            } else {
                const float* wb = wig + e * 81;
                float R[5][5];
                #pragma unroll
                for (int d = 0; d < 5; ++d)
                    #pragma unroll
                    for (int q = 0; q < 5; ++q) R[d][q] = wb[(4 + d) * 9 + 4 + q];
                for (int m = 0; m < 16; ++m) {
                    float c5[5], n5[5];
                    #pragma unroll
                    for (int d = 0; d < 5; ++d) {
                        c5[d] = cf[160 + m * 5 + d];
                        n5[d] = nf[160 + m * 5 + d];
                    }
                    #pragma unroll
                    for (int q = 0; q < 5; ++q) {
                        float cr = 0.f, nr = 0.f;
                        #pragma unroll
                        for (int d = 0; d < 5; ++d) {
                            cr += c5[d] * R[d][q];
                            nr += n5[d] * R[d][q];
                        }
                        PUTSIM(160 + m * 5 + q, cr * nr);
                    }
                }
            }
#undef PUTSIM
        }
    }
    __syncthreads();

    // ---------------- GEMM1: sim -> latent (silu) -> s_fused ----------------
    for (int ii = 0; ii < 3; ++ii) {
        // all scalar (wave-uniform) tile bookkeeping
        const int nt  = __builtin_amdgcn_readfirstlane(wave + ii * 8);   // 0..23
        const int l   = __builtin_amdgcn_readfirstlane(nt >> 3);
        const int ntl = __builtin_amdgcn_readfirstlane(nt & 7);
        const int nK  = __builtin_amdgcn_readfirstlane((l == 0) ? 2 : 3);
        const int bt  = __builtin_amdgcn_readfirstlane(
                            (l == 0) ? ntl * 2 : (l == 1 ? 16 + ntl * 3 : 40 + ntl * 3));
        const int ktb = __builtin_amdgcn_readfirstlane((l == 0) ? 0 : (l == 1 ? 2 : 5));
        v16h B[3];
        #pragma unroll
        for (int k = 0; k < 3; ++k)
            if (k < nK) B[k] = *(const v16h*)(wsB + ((bt + k) << 9) + (lane << 4));
        const float* bias = (l == 0) ? bs0 : ((l == 1) ? bs1 : bs2);
        const float bv    = bias[ntl * 16 + (lane & 15)];
        const int ncol = l * 128 + ntl * 16 + (lane & 15);   // fused K index
        const int kt = ncol >> 5, kk = ncol & 31;
        const int soff = (kk & 7) + ((kk >> 4) << 3);
        const int lhi  = ((kk >> 3) & 1) << 4;
        #pragma unroll
        for (int mt = 0; mt < 4; ++mt) {
            v8f acc = {};
            #pragma unroll
            for (int k = 0; k < 3; ++k) {
                if (k < nK) {
                    v16h a = *(const v16h*)(s_ovl + (((mt << 3) + ktb + k) << 9) + (lane << 4));
                    acc = __builtin_amdgcn_wmma_f32_16x16x32_f16(false, a, false, B[k],
                                                                 (short)0, acc, false, false);
                }
            }
            #pragma unroll
            for (int r = 0; r < 8; ++r) {
                float v = fast_silu(acc[r] + bv);
                int m = r + ((lane >> 4) << 3);
                s_fused[((mt * 12 + kt) << 9) + ((m + lhi) << 4) + soff] = (_Float16)v;
            }
        }
    }
    __syncthreads();

    // ---------------- GEMM2: fused -> h (silu) -> s_ovl ---------------------
    for (int ii = 0; ii < 2; ++ii) {
        const int nt = __builtin_amdgcn_readfirstlane(wave + ii * 8);    // 0..15
        v16h B[12];
        #pragma unroll
        for (int k = 0; k < 12; ++k)
            B[k] = *(const v16h*)(wsB + ((64 + nt * 12 + k) << 9) + (lane << 4));
        const float bv = bf1[nt * 16 + (lane & 15)];
        const int ncol = nt * 16 + (lane & 15);              // h K index
        const int kt = ncol >> 5, kk = ncol & 31;
        const int soff = (kk & 7) + ((kk >> 4) << 3);
        const int lhi  = ((kk >> 3) & 1) << 4;
        #pragma unroll
        for (int mt = 0; mt < 4; ++mt) {
            v8f acc = {};
            #pragma unroll
            for (int k = 0; k < 12; ++k) {
                v16h a = *(const v16h*)(s_fused + ((mt * 12 + k) << 9) + (lane << 4));
                acc = __builtin_amdgcn_wmma_f32_16x16x32_f16(false, a, false, B[k],
                                                             (short)0, acc, false, false);
            }
            #pragma unroll
            for (int r = 0; r < 8; ++r) {
                float v = fast_silu(acc[r] + bv);
                int m = r + ((lane >> 4) << 3);
                s_ovl[((mt * 8 + kt) << 9) + ((m + lhi) << 4) + soff] = (_Float16)v;
            }
        }
    }
    __syncthreads();

    // ---------------- GEMM3: h -> out ---------------------------------------
    {
        const int nt = wave;                 // 0..7, already scalar
        v16h B[8];
        #pragma unroll
        for (int k = 0; k < 8; ++k)
            B[k] = *(const v16h*)(wsB + ((256 + nt * 8 + k) << 9) + (lane << 4));
        const float bv = bf2[nt * 16 + (lane & 15)];
        const int n = lane & 15;
        #pragma unroll
        for (int mt = 0; mt < 4; ++mt) {
            v8f acc = {};
            #pragma unroll
            for (int k = 0; k < 8; ++k) {
                v16h a = *(const v16h*)(s_ovl + ((mt * 8 + k) << 9) + (lane << 4));
                acc = __builtin_amdgcn_wmma_f32_16x16x32_f16(false, a, false, B[k],
                                                             (short)0, acc, false, false);
            }
            #pragma unroll
            for (int r = 0; r < 8; ++r) {
                int mloc = mt * 16 + r + ((lane >> 4) << 3);
                long e = e0 + mloc;
                if (e < N_EDGES) out[e * 128 + nt * 16 + n] = acc[r] + bv;
            }
        }
    }
}

// ---------------------------------------------------------------------------
extern "C" void kernel_launch(void* const* d_in, const int* in_sizes, int n_in,
                              void* d_out, int out_size, void* d_ws, size_t ws_size,
                              hipStream_t stream) {
    const float* node_features = (const float*)d_in[0];
    // d_in[1] active_edge_vector is unused by the reference
    const int*   eidx = (const int*)d_in[2];
    const float* wig  = (const float*)d_in[3];
    const float* Wc0 = (const float*)d_in[4],  *bc0 = (const float*)d_in[5];
    const float* Wc1 = (const float*)d_in[6],  *Wc2 = (const float*)d_in[7];
    const float* Wn0 = (const float*)d_in[8],  *bn0 = (const float*)d_in[9];
    const float* Wn1 = (const float*)d_in[10], *Wn2 = (const float*)d_in[11];
    const float* Ws0 = (const float*)d_in[12], *bs0 = (const float*)d_in[13];
    const float* Ws1 = (const float*)d_in[14], *bs1 = (const float*)d_in[15];
    const float* Ws2 = (const float*)d_in[16], *bs2 = (const float*)d_in[17];
    const float* Wf1 = (const float*)d_in[18], *bf1 = (const float*)d_in[19];
    const float* Wf2 = (const float*)d_in[20], *bf2 = (const float*)d_in[21];
    float* out = (float*)d_out;

    // workspace layout: center rows | neigh rows | f16 B tiles
    float*    wsC = (float*)d_ws;
    float*    wsN = wsC + (size_t)N_NODES * FEAT;
    _Float16* wsB = (_Float16*)(wsN + (size_t)N_NODES * FEAT);

    prep_weights<<<(320 * 512 + 255) / 256, 256, 0, stream>>>(Ws0, Ws1, Ws2, Wf1, Wf2, wsB);

    long tot = (long)N_NODES * FEAT;
    node_linear<<<(int)((tot + 255) / 256), 256, 0, stream>>>(
        node_features, Wc0, bc0, Wc1, Wc2, Wn0, bn0, Wn1, Wn2, wsC, wsN);

    edge_kernel<<<(N_EDGES + 63) / 64, 256, 0, stream>>>(
        eidx, wig, wsC, wsN, wsB, bs0, bs1, bs2, bf1, bf2, out);
}
